// AttentionHead_36335423324709
// MI455X (gfx1250) — matrix operand order repaired and checked by reading, not compiled, not run on hardware
//
#include <hip/hip_runtime.h>
#include <hip/hip_bf16.h>

#define B_  4
#define S_  2048
#define E_  2048
#define DH_ 256

typedef __bf16 bf16_t;
typedef __attribute__((ext_vector_type(16))) __bf16 v16bf;
typedef __attribute__((ext_vector_type(8)))  __bf16 v8bf;
typedef __attribute__((ext_vector_type(4)))  __bf16 v4bf;
typedef __attribute__((ext_vector_type(8)))  float  v8f;

static __device__ inline v16bf cat8(v8bf lo, v8bf hi) {
  return __builtin_shufflevector(lo, hi, 0,1,2,3,4,5,6,7,8,9,10,11,12,13,14,15);
}

// A-operand (16x32, 16-bit): lane L holds row m = L&15.
// half = L>>4: lo 8 elems at k = kbase + 8*half, hi 8 at k = kbase + 16 + 8*half.
static __device__ inline v16bf load_a(const bf16_t* __restrict__ base, int stride,
                                      int mbase, int kbase, int lane) {
  const int m  = mbase + (lane & 15);
  const int hv = lane >> 4;
  const bf16_t* p = base + (size_t)m * stride + kbase + hv * 8;
  v8bf lo = *(const v8bf*)(p);
  v8bf hi = *(const v8bf*)(p + 16);
  return cat8(lo, hi);
}

// B-operand (32x16, 16-bit): lane L holds column n = L&15; half = L>>4:
// lo 8 elems at k = kbase + 8*half, hi 8 at k = kbase + 16 + 8*half. Storage must be k-contiguous per column
// (base = transposed matrix [n][k]).
static __device__ inline v16bf load_b(const bf16_t* __restrict__ base, int stride,
                                      int nbase, int kbase, int lane) {
  const int n  = nbase + (lane & 15);
  const int hv = lane >> 4;
  const bf16_t* p = base + (size_t)n * stride + kbase + hv * 8;
  v8bf lo = *(const v8bf*)(p);
  v8bf hi = *(const v8bf*)(p + 16);
  return cat8(lo, hi);
}

static __device__ inline v8f wmma_bf16(v16bf a, v16bf b, v8f c) {
  return __builtin_amdgcn_wmma_f32_16x16x32_bf16(false, a, false, b, (short)0, c, false, false);
}

// ---------------- conversion kernels ----------------

__global__ __launch_bounds__(256) void cvt_x_kernel(const float* __restrict__ x,
                                                    bf16_t* __restrict__ xbf, long n) {
  long i = ((long)blockIdx.x * blockDim.x + threadIdx.x) * 4;
  if (i < n) {
    float4 v = *(const float4*)(x + i);
    v4bf o = { (bf16_t)v.x, (bf16_t)v.y, (bf16_t)v.z, (bf16_t)v.w };
    *(v4bf*)(xbf + i) = o;
  }
}

// wt[mat][n][k] = W_mat[k][n]  (bf16, transposed so GEMM B-operand is k-contiguous)
__global__ __launch_bounds__(256) void cvt_w_kernel(const float* __restrict__ wq,
                                                    const float* __restrict__ wk,
                                                    const float* __restrict__ wv,
                                                    bf16_t* __restrict__ wt) {
  int tid = blockIdx.x * blockDim.x + threadIdx.x;   // over 3*E*DH
  int mat = tid / (E_ * DH_);
  int r   = tid % (E_ * DH_);
  int k   = r / DH_;
  int n   = r % DH_;
  const float* w = (mat == 0) ? wq : (mat == 1) ? wk : wv;
  wt[(size_t)mat * DH_ * E_ + (size_t)n * E_ + k] = (bf16_t)w[r];
}

// ---------------- projection GEMM: Q,K,V = x @ W + b ----------------
// One wave per 32x64 output tile (2 A-ops x 4 B-ops -> 8 WMMA per k-step).
// Q,K stored bf16 [B*S][DH]; V stored TRANSPOSED bf16 [B][DH][S] so the
// attention P@V B-operand is key-contiguous.

__global__ __launch_bounds__(256) void proj_kernel(const bf16_t* __restrict__ xbf,
                                                   const bf16_t* __restrict__ wt,
                                                   const float* __restrict__ bQ,
                                                   const float* __restrict__ bK,
                                                   const float* __restrict__ bV,
                                                   bf16_t* __restrict__ qbf,
                                                   bf16_t* __restrict__ kbf,
                                                   bf16_t* __restrict__ vt) {
  const int lane = threadIdx.x & 31;
  const int widx = blockIdx.x * (blockDim.x >> 5) + (threadIdx.x >> 5);
  const int tiles_per_mat = (B_ * S_ / 32) * (DH_ / 64);   // 256 * 4 = 1024
  const int mat   = widx / tiles_per_mat;
  const int r     = widx % tiles_per_mat;
  const int mbase = (r / (DH_ / 64)) * 32;
  const int nbase = (r % (DH_ / 64)) * 64;

  const bf16_t* wmat = wt + (size_t)mat * DH_ * E_;
  v8f acc[2][4];
#pragma unroll
  for (int i = 0; i < 2; ++i)
#pragma unroll
    for (int j = 0; j < 4; ++j) acc[i][j] = (v8f){};

#pragma unroll 2
  for (int k0 = 0; k0 < E_; k0 += 32) {
    v16bf a0 = load_a(xbf, E_, mbase,      k0, lane);
    v16bf a1 = load_a(xbf, E_, mbase + 16, k0, lane);
    v16bf b0 = load_b(wmat, E_, nbase,      k0, lane);
    v16bf b1 = load_b(wmat, E_, nbase + 16, k0, lane);
    v16bf b2 = load_b(wmat, E_, nbase + 32, k0, lane);
    v16bf b3 = load_b(wmat, E_, nbase + 48, k0, lane);
    acc[0][0] = wmma_bf16(a0, b0, acc[0][0]);
    acc[0][1] = wmma_bf16(a0, b1, acc[0][1]);
    acc[0][2] = wmma_bf16(a0, b2, acc[0][2]);
    acc[0][3] = wmma_bf16(a0, b3, acc[0][3]);
    acc[1][0] = wmma_bf16(a1, b0, acc[1][0]);
    acc[1][1] = wmma_bf16(a1, b1, acc[1][1]);
    acc[1][2] = wmma_bf16(a1, b2, acc[1][2]);
    acc[1][3] = wmma_bf16(a1, b3, acc[1][3]);
  }

  const int hv  = lane >> 4;
  const int l16 = lane & 15;
  const float* bp = (mat == 0) ? bQ : (mat == 1) ? bK : bV;
#pragma unroll
  for (int j = 0; j < 4; ++j) {
    const int n = nbase + j * 16 + l16;
    const float bias = bp[n];
#pragma unroll
    for (int i = 0; i < 2; ++i) {
#pragma unroll
      for (int e = 0; e < 8; ++e) {
        const float v = acc[i][j][e] + bias;
        const int m = mbase + i * 16 + e + hv * 8;
        if (mat == 0) {
          qbf[(size_t)m * DH_ + n] = (bf16_t)v;
        } else if (mat == 1) {
          kbf[(size_t)m * DH_ + n] = (bf16_t)v;
        } else {
          const int bb = m / S_, s = m % S_;
          vt[((size_t)bb * DH_ + n) * S_ + s] = (bf16_t)v;
        }
      }
    }
  }
}

// ---------------- flash attention: one wave per 16-query tile ----------------

__global__ __launch_bounds__(32) void attn_kernel(const bf16_t* __restrict__ qbf,
                                                  const bf16_t* __restrict__ kbf,
                                                  const bf16_t* __restrict__ vt,
                                                  const int* __restrict__ pad,
                                                  float* __restrict__ out) {
  __shared__ __attribute__((aligned(16))) bf16_t pbuf[16 * 32];   // P tile staging

  const int lane  = threadIdx.x;
  const int b     = blockIdx.x / (S_ / 16);
  const int qbase = (blockIdx.x % (S_ / 16)) * 16;
  const int hv    = lane >> 4;
  const int l16   = lane & 15;

  const bf16_t* qmat = qbf + (size_t)b * S_ * DH_;
  const bf16_t* kmat = kbf + (size_t)b * S_ * DH_;
  const bf16_t* vmat = vt  + (size_t)b * DH_ * S_;
  const int*    padb = pad + (size_t)b * S_;

  // Q tile resident in registers for the whole key loop (8 x v16bf = 64 VGPRs).
  v16bf qa[8];
#pragma unroll
  for (int d = 0; d < 8; ++d) qa[d] = load_a(qmat, DH_, qbase, d * 32, lane);

  float m_run[8], l_run[8];
  v8f O[16];
#pragma unroll
  for (int i = 0; i < 8; ++i) { m_run[i] = -__builtin_inff(); l_run[i] = 0.f; }
#pragma unroll
  for (int t = 0; t < 16; ++t) O[t] = (v8f){};

  const float scale = 0.0625f;   // 1/sqrt(256)
  const int kend = qbase + 16;   // causal: no key block beyond the query tile

  for (int kb = 0; kb < kend; kb += 32) {
    // Prefetch next key block of K and (a slice of) V into cache (global_prefetch_b8).
    if (kb + 32 < kend) {
      __builtin_prefetch(kmat + (size_t)(kb + 32 + lane) * DH_, 0, 1);
      __builtin_prefetch(vmat + (size_t)(lane * 8) * S_ + kb + 32, 0, 1);
    }

    // ---- scores: two 16x16 tiles over keys [kb, kb+32) ----
    v8f s0 = {}, s1 = {};
#pragma unroll
    for (int d = 0; d < 8; ++d) {
      v16bf b0 = load_b(kmat, DH_, kb,      d * 32, lane);
      v16bf b1 = load_b(kmat, DH_, kb + 16, d * 32, lane);
      s0 = wmma_bf16(qa[d], b0, s0);
      s1 = wmma_bf16(qa[d], b1, s1);
    }

    // ---- mask + scale in C/D layout: row = i + 8*hv, col key = kb(+16)+l16 ----
    const int key0 = kb + l16;
    const int key1 = kb + 16 + l16;
    const int p0 = padb[key0];
    const int p1 = padb[key1];
#pragma unroll
    for (int i = 0; i < 8; ++i) {
      const int q = qbase + i + hv * 8;
      const bool ok0 = (key0 <= q) && (p0 == 0);
      const bool ok1 = (key1 <= q) && (p1 == 0);
      s0[i] = ok0 ? s0[i] * scale : -__builtin_inff();
      s1[i] = ok1 ? s1[i] * scale : -__builtin_inff();
    }

    // ---- online softmax: row reductions within 16-lane halves ----
    float alpha[8];
#pragma unroll
    for (int i = 0; i < 8; ++i) {
      float v = fmaxf(s0[i], s1[i]);
      v = fmaxf(v, __shfl_xor(v, 1));
      v = fmaxf(v, __shfl_xor(v, 2));
      v = fmaxf(v, __shfl_xor(v, 4));
      v = fmaxf(v, __shfl_xor(v, 8));
      const float mnew = fmaxf(m_run[i], v);
      alpha[i] = __expf(m_run[i] - mnew);
      m_run[i] = mnew;
      const float e0 = __expf(s0[i] - mnew);
      const float e1 = __expf(s1[i] - mnew);
      s0[i] = e0;
      s1[i] = e1;
      float rs = e0 + e1;
      rs += __shfl_xor(rs, 1);
      rs += __shfl_xor(rs, 2);
      rs += __shfl_xor(rs, 4);
      rs += __shfl_xor(rs, 8);
      l_run[i] = l_run[i] * alpha[i] + rs;
    }

    // ---- P: C/D layout -> bf16 LDS [16 rows][32 keys] -> reload in A layout ----
    __syncthreads();
#pragma unroll
    for (int i = 0; i < 8; ++i) {
      const int m = i + hv * 8;
      pbuf[m * 32 + l16]      = (bf16_t)s0[i];
      pbuf[m * 32 + 16 + l16] = (bf16_t)s1[i];
    }
    __syncthreads();
    v16bf pa;
    {
      const bf16_t* p = &pbuf[l16 * 32 + hv * 8];
      v8bf lo = *(const v8bf*)(p);
      v8bf hi = *(const v8bf*)(p + 16);
      pa = cat8(lo, hi);
    }

    // ---- rescale O, accumulate P @ V over the 32-key block ----
#pragma unroll
    for (int t = 0; t < 16; ++t)
#pragma unroll
      for (int i = 0; i < 8; ++i) O[t][i] *= alpha[i];
#pragma unroll
    for (int t = 0; t < 16; ++t) {
      v16bf bv = load_b(vmat, S_, t * 16, kb, lane);   // Vt rows: d contiguous in key
      O[t] = wmma_bf16(pa, bv, O[t]);
    }
  }

  // ---- normalize and write fp32 output ----
  float inv[8];
#pragma unroll
  for (int i = 0; i < 8; ++i) inv[i] = 1.0f / l_run[i];
#pragma unroll
  for (int t = 0; t < 16; ++t) {
#pragma unroll
    for (int i = 0; i < 8; ++i) {
      const int q = qbase + i + hv * 8;
      const int d = t * 16 + l16;
      out[((size_t)b * S_ + q) * DH_ + d] = O[t][i] * inv[i];
    }
  }
}

// ---------------- host launch ----------------

extern "C" void kernel_launch(void* const* d_in, const int* in_sizes, int n_in,
                              void* d_out, int out_size, void* d_ws, size_t ws_size,
                              hipStream_t stream) {
  const float* x   = (const float*)d_in[0];
  // d_in[1] = causal_mask (unused: causality applied structurally)
  const int*   pad = (const int*)d_in[2];
  const float* WQ  = (const float*)d_in[3];
  const float* bQ  = (const float*)d_in[4];
  const float* WK  = (const float*)d_in[5];
  const float* bK  = (const float*)d_in[6];
  const float* WV  = (const float*)d_in[7];
  const float* bV  = (const float*)d_in[8];
  float* out = (float*)d_out;

  char* ws = (char*)d_ws;
  size_t off = 0;
  bf16_t* xbf = (bf16_t*)(ws + off); off += (size_t)B_ * S_ * E_ * 2;      // 33.5 MB
  bf16_t* wt  = (bf16_t*)(ws + off); off += (size_t)3 * DH_ * E_ * 2;      //  3.1 MB
  bf16_t* qbf = (bf16_t*)(ws + off); off += (size_t)B_ * S_ * DH_ * 2;     //  4.2 MB
  bf16_t* kbf = (bf16_t*)(ws + off); off += (size_t)B_ * S_ * DH_ * 2;     //  4.2 MB
  bf16_t* vt  = (bf16_t*)(ws + off); off += (size_t)B_ * DH_ * S_ * 2;     //  4.2 MB

  const long nx = (long)B_ * S_ * E_;
  cvt_x_kernel<<<(int)(nx / 4 / 256), 256, 0, stream>>>(x, xbf, nx);
  cvt_w_kernel<<<(3 * E_ * DH_) / 256, 256, 0, stream>>>(WQ, WK, WV, wt);

  const int total_waves = 3 * (B_ * S_ / 32) * (DH_ / 64);   // 3072 waves
  proj_kernel<<<total_waves / 8, 256, 0, stream>>>(xbf, wt, bQ, bK, bV, qbf, kbf, vt);

  attn_kernel<<<B_ * (S_ / 16), 32, 0, stream>>>(qbf, kbf, vt, pad, out);
}
